// SpGAT_44504451121554
// MI455X (gfx1250) — matrix operand order repaired and checked by reading, not compiled, not run on hardware
//
#include <hip/hip_runtime.h>
#include <hip/hip_bf16.h>

#define GAT_LRELU_ALPHA 0.2f

typedef __attribute__((ext_vector_type(16))) __bf16 v16bf;
typedef __attribute__((ext_vector_type(8)))  float  v8f;

// ---------------- f32 -> bf16 conversions ----------------
__global__ __launch_bounds__(256)
void k_f32_to_bf16(const float* __restrict__ in, __bf16* __restrict__ out, long n) {
    long t = (long)blockIdx.x * blockDim.x + threadIdx.x;
    if (t < n) out[t] = (__bf16)in[t];
}

// in: [Z][K][Nc] f32  ->  out: [Z][Nc][K] bf16   (transpose so B-fragments are contiguous)
__global__ __launch_bounds__(256)
void k_f32_to_bf16_transpose(const float* __restrict__ in, __bf16* __restrict__ out,
                             int K, int Nc, long total) {
    long t = (long)blockIdx.x * blockDim.x + threadIdx.x;
    if (t >= total) return;
    long plane = (long)K * Nc;
    long z = t / plane;
    long r = t - z * plane;
    int  k = (int)(r / Nc);
    int  n = (int)(r - (long)k * Nc);
    out[z * plane + (long)n * K + k] = (__bf16)in[t];
}

// ---------------- bf16 WMMA GEMM, register-blocked ----------------
// One wave computes a 16 x (16*NT) strip of C: the A fragment is loaded once per
// k-step and reused across NT B fragments / NT f32 accumulators.
//   A  : [M x K] bf16 row-major
//   Bt : [Z][Ncols x K] bf16 (B transposed, so each B fragment is 2 contiguous b128s)
//   C  : [M x ldc] f32, per-z column offset = z*cColPerZ
template <int NT>
__global__ __launch_bounds__(32)
void k_gemm_bf16_wmma(const __bf16* __restrict__ A,
                      const __bf16* __restrict__ Bt,
                      float* __restrict__ C,
                      int M, int K, int ldc, long bStrideZ, int cColPerZ) {
    const int lane = threadIdx.x & 31;
    const int m0 = blockIdx.x * 16;
    const int n0 = blockIdx.y * (16 * NT);
    const int z  = blockIdx.z;

    // A fragment (16x32 bf16, ISA layout): lanes 0-15 row m0+l, K {0..7,16..23};
    // lanes 16-31 same rows, K {8..15,24..31}
    int arow = m0 + (lane & 15);
    if (arow >= M) arow = M - 1;          // uniform-EXEC clamp (stores are guarded)
    const int khalfA = (lane >> 4) * 8;
    const __bf16* __restrict__ arowp = A + (long)arow * K;

    // B fragment (32x16 bf16): lane&15 = column; lanes 0-15 K {0..7,16..23}, lanes 16-31 K {8..15,24..31}
    const int krowB = (lane >> 4) * 8;
    const __bf16* __restrict__ bbase =
        Bt + (long)z * bStrideZ + (long)(n0 + (lane & 15)) * K + krowB;

    v8f acc[NT] = {};
    for (int k0 = 0; k0 < K; k0 += 32) {
        v16bf a;
        __builtin_memcpy(&a,             arowp + k0 + khalfA,      16);  // K khalf..+7
        __builtin_memcpy((char*)&a + 16, arowp + k0 + khalfA + 16, 16);  // K 16+khalf..+7
        #pragma unroll
        for (int j = 0; j < NT; ++j) {
            v16bf b;
            const __bf16* bp = bbase + (long)j * 16 * K + k0;
            __builtin_memcpy(&b,             bp,     16);                // K krowB..+7
            __builtin_memcpy((char*)&b + 16, bp + 16, 16);               // K krowB+16..+23
            acc[j] = __builtin_amdgcn_wmma_f32_16x16x32_bf16(
                         /*neg_a=*/false, a, /*neg_b=*/false, b,
                         /*c_mod=*/(short)0, acc[j],
                         /*reuse_a=*/false, /*reuse_b=*/false);
        }
    }

    // C layout: VGPR r -> lanes 0-15: M=m0+r, lanes 16-31: M=m0+8+r; N = +(lane&15)
    const int rowBase = m0 + (lane >> 4) * 8;
    #pragma unroll
    for (int j = 0; j < NT; ++j) {
        const int ccol = z * cColPerZ + n0 + j * 16 + (lane & 15);
        #pragma unroll
        for (int r = 0; r < 8; ++r) {
            int row = rowBase + r;
            if (row < M) C[(long)row * ldc + ccol] = acc[j][r];
        }
    }
}

// ---------------- per-node attention scores: sd = H . a[:F], ss = H . a[F:] ----------------
__global__ __launch_bounds__(256)
void k_scores(const float* __restrict__ H,   // [N][nheads*F]
              const float* __restrict__ a,   // [nheads][2F]
              float* __restrict__ sd, float* __restrict__ ss,
              int n_nodes, int nheads, int F) {
    int t = blockIdx.x * blockDim.x + threadIdx.x;
    if (t >= n_nodes * nheads) return;
    int n = t / nheads, h = t - n * nheads;
    const float* hv = H + (long)n * nheads * F + (long)h * F;
    const float* ad = a + (long)h * 2 * F;
    const float* as = ad + F;
    float accd = 0.f, accs = 0.f;
    for (int f = 0; f < F; ++f) {
        float xv = hv[f];
        accd += xv * ad[f];
        accs += xv * as[f];
    }
    sd[t] = accd;
    ss[t] = accs;
}

// ---------------- wave-per-edge: e = exp(-lrelu(sd[d]+ss[s])); atomic accumulate ----------------
__global__ __launch_bounds__(256)
void k_edge(const int* __restrict__ src, const int* __restrict__ dst,
            const float* __restrict__ sd, const float* __restrict__ ss,
            const float* __restrict__ H,          // [N][nheads*F]
            float* __restrict__ rowsum,           // [N][nheads]
            float* __restrict__ num,              // [N][nheads*F]
            int n_edges, int nheads, int F) {
    long gid = (long)blockIdx.x * blockDim.x + threadIdx.x;
    int e = (int)(gid >> 5);
    int lane = (int)(gid & 31);
    if (e >= n_edges) return;
    int s = src[e], d = dst[e];
    const float* __restrict__ hs = H   + (long)s * nheads * F;
    float*       __restrict__ nm = num + (long)d * nheads * F;
    for (int h = 0; h < nheads; ++h) {
        float sc = sd[(long)d * nheads + h] + ss[(long)s * nheads + h];
        float lr = sc > 0.f ? sc : GAT_LRELU_ALPHA * sc;
        float ev = __expf(-lr);
        if (lane == 0) atomicAdd(&rowsum[(long)d * nheads + h], ev);
        for (int f = lane; f < F; f += 32)
            atomicAdd(&nm[h * F + f], ev * hs[h * F + f]);
    }
}

// ---------------- normalize + ELU, emit bf16 activations for layer-2 GEMM ----------------
__global__ __launch_bounds__(256)
void k_norm_elu_bf16(const float* __restrict__ num, const float* __restrict__ rowsum,
                     __bf16* __restrict__ out, long total, int nheads, int F) {
    long t = (long)blockIdx.x * blockDim.x + threadIdx.x;
    if (t >= total) return;
    int hf = nheads * F;
    long n = t / hf;
    int  j = (int)(t - n * hf);
    int  h = j / F;
    float r = rowsum[n * nheads + h];
    float v = num[t] / (r == 0.f ? 1.f : r);
    v = v > 0.f ? v : (__expf(v) - 1.f);
    out[t] = (__bf16)v;
}

// ---------------- final: isolated-node passthrough + ELU ----------------
__global__ __launch_bounds__(256)
void k_final(const float* __restrict__ num, const float* __restrict__ rowsum,
             const float* __restrict__ x, float* __restrict__ out,
             int n_nodes, int F) {
    long t = (long)blockIdx.x * blockDim.x + threadIdx.x;
    if (t >= (long)n_nodes * F) return;
    long n = t / F;
    float r = rowsum[n];                         // rowsum==0  <=>  degree==0 (e>0 always)
    float v = (r == 0.f) ? x[t] : num[t] / r;
    out[t] = v > 0.f ? v : (__expf(v) - 1.f);
}

extern "C" void kernel_launch(void* const* d_in, const int* in_sizes, int n_in,
                              void* d_out, int out_size, void* d_ws, size_t ws_size,
                              hipStream_t stream) {
    const int NFEAT = 128, NHID = 64, NHEADS = 8, NOUT = 128;
    const float* x  = (const float*)d_in[0];
    const int*   src = (const int*)d_in[1];
    const int*   dst = (const int*)d_in[2];
    const float* W1 = (const float*)d_in[3];
    const float* a1 = (const float*)d_in[4];
    const float* W2 = (const float*)d_in[5];
    const float* a2 = (const float*)d_in[6];
    float* out = (float*)d_out;

    const int Nn = in_sizes[0] / NFEAT;   // 50000
    const int E  = in_sizes[1];           // 800000
    const int F1 = NHEADS * NHID;         // 512

    // ---- carve workspace (256B aligned) ----
    char* p = (char*)d_ws;
    auto take = [&](size_t bytes) { char* r = p; p += (bytes + 255) & ~(size_t)255; return r; };
    __bf16* xb   = (__bf16*)take((size_t)Nn * NFEAT * 2);
    __bf16* W1bt = (__bf16*)take((size_t)NHEADS * NFEAT * NHID * 2);   // [H][64][128]
    __bf16* W2bt = (__bf16*)take((size_t)F1 * NOUT * 2);               // [128][512]
    float*  H1   = (float*) take((size_t)Nn * F1 * 4);                 // [N][H*64]
    float*  sd1  = (float*) take((size_t)Nn * NHEADS * 4);
    float*  ss1  = (float*) take((size_t)Nn * NHEADS * 4);
    __bf16* x2b  = (__bf16*)take((size_t)Nn * F1 * 2);                 // elu'd concat, bf16
    float*  H2   = (float*) take((size_t)Nn * NOUT * 4);
    float*  sd2  = (float*) take((size_t)Nn * 4);
    float*  ss2  = (float*) take((size_t)Nn * 4);
    char*   zbase = p;                                                  // zeroed every call
    float*  rowsum1 = (float*)take((size_t)Nn * NHEADS * 4);
    float*  num1    = (float*)take((size_t)Nn * F1 * 4);
    float*  rowsum2 = (float*)take((size_t)Nn * 4);
    float*  num2    = (float*)take((size_t)Nn * NOUT * 4);
    size_t  zbytes = (size_t)(p - zbase);
    hipMemsetAsync(zbase, 0, zbytes, stream);   // graph-capture-safe stream memset

    auto cdiv = [](long a, long b) { return (unsigned)((a + b - 1) / b); };

    // ---- precision conversion / weight transpose ----
    k_f32_to_bf16<<<cdiv((long)Nn * NFEAT, 256), 256, 0, stream>>>(x, xb, (long)Nn * NFEAT);
    k_f32_to_bf16_transpose<<<cdiv((long)NHEADS * NFEAT * NHID, 256), 256, 0, stream>>>(
        W1, W1bt, NFEAT, NHID, (long)NHEADS * NFEAT * NHID);
    k_f32_to_bf16_transpose<<<cdiv((long)F1 * NOUT, 256), 256, 0, stream>>>(
        W2, W2bt, F1, NOUT, (long)F1 * NOUT);

    // ---- layer 1: H1[n][head*64 + c], one wave covers all 64 hidden cols (NT=4) ----
    dim3 g1(cdiv(Nn, 16), 1, NHEADS);
    k_gemm_bf16_wmma<4><<<g1, 32, 0, stream>>>(xb, W1bt, H1, Nn, NFEAT, F1,
                                               (long)NHID * NFEAT, NHID);
    k_scores<<<cdiv((long)Nn * NHEADS, 256), 256, 0, stream>>>(H1, a1, sd1, ss1, Nn, NHEADS, NHID);
    k_edge<<<cdiv((long)E * 32, 256), 256, 0, stream>>>(src, dst, sd1, ss1, H1,
                                                        rowsum1, num1, E, NHEADS, NHID);
    k_norm_elu_bf16<<<cdiv((long)Nn * F1, 256), 256, 0, stream>>>(num1, rowsum1, x2b,
                                                                  (long)Nn * F1, NHEADS, NHID);

    // ---- layer 2: one wave covers all 128 output cols (NT=8) ----
    dim3 g2(cdiv(Nn, 16), 1, 1);
    k_gemm_bf16_wmma<8><<<g2, 32, 0, stream>>>(x2b, W2bt, H2, Nn, F1, NOUT, 0, 0);
    k_scores<<<cdiv(Nn, 256), 256, 0, stream>>>(H2, a2, sd2, ss2, Nn, 1, NOUT);
    k_edge<<<cdiv((long)E * 32, 256), 256, 0, stream>>>(src, dst, sd2, ss2, H2,
                                                        rowsum2, num2, E, 1, NOUT);
    k_final<<<cdiv((long)Nn * NOUT, 256), 256, 0, stream>>>(num2, rowsum2, x, out, Nn, NOUT);
}
